// ChessformerAttention_39127152067292
// MI455X (gfx1250) — compile-verified, matrix operand order checked by reading, not run on hardware
//
#include <hip/hip_runtime.h>
#include <hip/hip_bf16.h>

typedef __bf16 v16bf  __attribute__((ext_vector_type(16)));
typedef __bf16 bf16x4 __attribute__((ext_vector_type(4)));
typedef float  v8f    __attribute__((ext_vector_type(8)));

// ---------------------------------------------------------------------------
// Shapes
// ---------------------------------------------------------------------------
static constexpr int Bn = 256, Ln = 64, Dn = 1024, Hn = 16, HDn = 64;
static constexpr int Mtot = Bn * Ln;         // 16384 rows of X

// ---------------------------------------------------------------------------
// WMMA helpers
// ---------------------------------------------------------------------------
__device__ __forceinline__ v8f zero8() {
  v8f z = {0.f, 0.f, 0.f, 0.f, 0.f, 0.f, 0.f, 0.f};
  return z;
}

__device__ __forceinline__ v8f wmma_bf16(v16bf a, v16bf b, v8f c) {
  // D = A(16x32) x B(32x16) + C(16x16), fp32 accumulate
  return __builtin_amdgcn_wmma_f32_16x16x32_bf16(
      /*neg_a=*/false, a, /*neg_b=*/false, b,
      /*c_mod=*/(short)0, c, /*reuse_a=*/false, /*reuse_b=*/false);
}

// Load a 16x32 bf16 fragment in the CDNA5 A-matrix VGPR layout from LDS.
// Element (m,k) lives at p[m*ld + k]. Per ISA table (16-bit A 16x32):
//   lane 0-15: M = lane, K-group low;  lane 16-31: M = lane-16, K +8
//   VGPR v (pair of bf16): Kbase = 2v (v<4) or 2v+8 (v>=4), +8*(lane>=16)
// B fragments use the same loader applied to B^T (n as the "row" index).
__device__ __forceinline__ v16bf load_frag(const __bf16* __restrict__ p,
                                           int ld, int lane) {
  const int m    = lane & 15;
  const int half = (lane >> 4) & 1;
  const __bf16* r = p + m * ld + 8 * half;
  v16bf f;
#pragma unroll
  for (int v = 0; v < 8; ++v) {
    const int kb = (v < 4) ? (2 * v) : (2 * v + 8);
    f[2 * v]     = r[kb];
    f[2 * v + 1] = r[kb + 1];
  }
  return f;
}

// ---------------------------------------------------------------------------
// Kernel 1: fused QKV projection.
//   [16384 x 1024] fp32  @  Wq|Wk|Wv [1024 x 1024] fp32  -> bf16 [B,H,L,HD]
// grid = (24, 128): x-> {which 0..2} x {8 N-tiles of 128}, y-> 128 M-tiles
// ---------------------------------------------------------------------------
__global__ __launch_bounds__(256) void qkv_kernel(
    const float* __restrict__ X,  const float* __restrict__ Wq,
    const float* __restrict__ Wk, const float* __restrict__ Wv,
    __bf16* __restrict__ qo, __bf16* __restrict__ ko, __bf16* __restrict__ vo) {
  __shared__ __bf16 lds_a[128 * 34];   // [row 0..127][k 0..31], pad 2
  __shared__ __bf16 lds_b[128 * 34];   // B^T: [n 0..127][k 0..31]

  const int tid    = threadIdx.x;
  const int lane   = tid & 31;
  const int wv     = tid >> 5;
  const int wave_m = wv & 3;    // 4 waves over M: 32 rows each
  const int wave_n = wv >> 2;   // 2 waves over N: 64 cols each

  const int which = blockIdx.x >> 3;            // 0=Q, 1=K, 2=V
  const int tileN = (blockIdx.x & 7) * 128;     // within 1024
  const int tileM = blockIdx.y * 128;

  const float* W = (which == 0) ? Wq : (which == 1) ? Wk : Wv;
  __bf16*      O = (which == 0) ? qo : (which == 1) ? ko : vo;

  v8f acc[2][4];
#pragma unroll
  for (int tm = 0; tm < 2; ++tm)
#pragma unroll
    for (int tn = 0; tn < 4; ++tn) acc[tm][tn] = zero8();

  for (int kb = 0; kb < 32; ++kb) {
    if (kb + 1 < 32)   // gfx1250 global_prefetch_b8 for next A slab
      __builtin_prefetch(X + (size_t)(tileM + (tid >> 3)) * Dn + (kb + 1) * 32, 0, 0);

    // Stage A: 128x32 fp32 -> bf16
#pragma unroll
    for (int i = 0; i < 4; ++i) {
      const int s   = tid + 256 * i;      // 0..1023
      const int row = s >> 3;
      const int c4  = (s & 7) << 2;
      const float4 f =
          *(const float4*)(X + (size_t)(tileM + row) * Dn + kb * 32 + c4);
      __bf16* d = &lds_a[row * 34 + c4];
      d[0] = (__bf16)f.x; d[1] = (__bf16)f.y;
      d[2] = (__bf16)f.z; d[3] = (__bf16)f.w;
    }
    // Stage B transposed: W[k][n] -> lds_b[n][k]
#pragma unroll
    for (int i = 0; i < 4; ++i) {
      const int s  = tid + 256 * i;
      const int k  = s >> 5;
      const int n4 = (s & 31) << 2;
      const float4 f =
          *(const float4*)(W + (size_t)(kb * 32 + k) * Dn + tileN + n4);
      lds_b[(n4 + 0) * 34 + k] = (__bf16)f.x;
      lds_b[(n4 + 1) * 34 + k] = (__bf16)f.y;
      lds_b[(n4 + 2) * 34 + k] = (__bf16)f.z;
      lds_b[(n4 + 3) * 34 + k] = (__bf16)f.w;
    }
    __syncthreads();

    const v16bf a0 = load_frag(&lds_a[(wave_m * 32 + 0)  * 34], 34, lane);
    const v16bf a1 = load_frag(&lds_a[(wave_m * 32 + 16) * 34], 34, lane);
#pragma unroll
    for (int tn = 0; tn < 4; ++tn) {
      const v16bf b = load_frag(&lds_b[(wave_n * 64 + tn * 16) * 34], 34, lane);
      acc[0][tn] = wmma_bf16(a0, b, acc[0][tn]);
      acc[1][tn] = wmma_bf16(a1, b, acc[1][tn]);
    }
    __syncthreads();
  }

  // Epilogue: scatter into [B,H,L,HD] bf16
  const int half = (lane >> 4) & 1;
  const int n0   = lane & 15;
#pragma unroll
  for (int tm = 0; tm < 2; ++tm)
#pragma unroll
    for (int tn = 0; tn < 4; ++tn)
#pragma unroll
      for (int r = 0; r < 8; ++r) {
        const int m  = tileM + wave_m * 32 + tm * 16 + r + 8 * half;
        const int n  = tileN + wave_n * 64 + tn * 16 + n0;
        const int bb = m >> 6, l = m & 63;
        const int h  = n >> 6, hd = n & 63;
        O[(((size_t)(bb * Hn + h) * Ln) + l) * HDn + hd] = (__bf16)acc[tm][tn][r];
      }
}

// ---------------------------------------------------------------------------
// Kernel 2: attention for one (b,h): S = QK^T/8 + bias; softmax; O = P V
// grid = B*H = 4096 blocks, 256 threads (8 waves; each wave owns 2 of the
// 16 16x16 tiles of the 64x64 matrices: same tm, adjacent tn).
// ---------------------------------------------------------------------------
__global__ __launch_bounds__(256) void attn_kernel(
    const __bf16* __restrict__ Q, const __bf16* __restrict__ K,
    const __bf16* __restrict__ V, const float* __restrict__ bias,
    __bf16* __restrict__ AO) {
  __shared__ __bf16 ldsQ [64 * 68];    // [l][hd], pad 4 -> 8B-aligned rows
  __shared__ __bf16 ldsK [64 * 68];    // [key][hd]
  __shared__ __bf16 ldsVt[64 * 68];    // transposed: [hd][key]
  __shared__ __bf16 ldsP [64 * 68];    // softmax probs, bf16
  __shared__ float  ldsS [64 * 65];    // logits / exp scratch

  const int tid  = threadIdx.x;
  const int lane = tid & 31;
  const int wv   = tid >> 5;
  const int bb   = blockIdx.x >> 4;
  const int h    = blockIdx.x & 15;
  const size_t base = (size_t)(bb * Hn + h) * Ln * HDn;

  // Stage Q, K, V (V transposed) into LDS
#pragma unroll
  for (int i = 0; i < 4; ++i) {
    const int s   = tid + 256 * i;    // 0..1023
    const int row = s >> 4;
    const int c4  = (s & 15) << 2;
    const bf16x4 q4 = *(const bf16x4*)(Q + base + row * HDn + c4);
    const bf16x4 k4 = *(const bf16x4*)(K + base + row * HDn + c4);
    const bf16x4 v4 = *(const bf16x4*)(V + base + row * HDn + c4);
    *(bf16x4*)(&ldsQ[row * 68 + c4]) = q4;
    *(bf16x4*)(&ldsK[row * 68 + c4]) = k4;
    ldsVt[(c4 + 0) * 68 + row] = v4[0];
    ldsVt[(c4 + 1) * 68 + row] = v4[1];
    ldsVt[(c4 + 2) * 68 + row] = v4[2];
    ldsVt[(c4 + 3) * 68 + row] = v4[3];
  }
  __syncthreads();

  const int tm  = wv >> 1;          // 0..3 (query row tile)
  const int tn0 = (wv & 1) * 2;     // 0 or 2 (key col tiles tn0, tn0+1)
  const int half = (lane >> 4) & 1;
  const int n0   = lane & 15;

  // ---- S = Q K^T (contraction over HD=64 -> two K=32 WMMA steps) ----
  v8f s0 = zero8(), s1 = zero8();
#pragma unroll
  for (int ks = 0; ks < 64; ks += 32) {
    const v16bf aq = load_frag(&ldsQ[tm * 16 * 68 + ks], 68, lane);
    const v16bf b0 = load_frag(&ldsK[tn0 * 16 * 68 + ks], 68, lane);
    const v16bf b1 = load_frag(&ldsK[(tn0 + 1) * 16 * 68 + ks], 68, lane);
    s0 = wmma_bf16(aq, b0, s0);
    s1 = wmma_bf16(aq, b1, s1);
  }
#pragma unroll
  for (int r = 0; r < 8; ++r) {
    const int m = tm * 16 + r + 8 * half;
    const int n = tn0 * 16 + n0;
    const float* brow = bias + ((size_t)h * Ln + m) * Ln;
    ldsS[m * 65 + n]      = s0[r] * 0.125f + brow[n];
    ldsS[m * 65 + n + 16] = s1[r] * 0.125f + brow[n + 16];
  }
  __syncthreads();

  // ---- row softmax (64 rows, one thread per row) ----
  if (tid < 64) {
    float mx = -3.4e38f;
#pragma unroll
    for (int j = 0; j < 64; ++j) mx = fmaxf(mx, ldsS[tid * 65 + j]);
    float sum = 0.f;
#pragma unroll
    for (int j = 0; j < 64; ++j) {
      const float e = __expf(ldsS[tid * 65 + j] - mx);
      sum += e;
      ldsS[tid * 65 + j] = e;
    }
    const float inv = 1.f / sum;
#pragma unroll
    for (int j = 0; j < 64; ++j)
      ldsP[tid * 68 + j] = (__bf16)(ldsS[tid * 65 + j] * inv);
  }
  __syncthreads();

  // ---- O = P V (contraction over key=64; V pre-transposed to [hd][key]) ----
  v8f o0 = zero8(), o1 = zero8();
#pragma unroll
  for (int ks = 0; ks < 64; ks += 32) {
    const v16bf ap = load_frag(&ldsP[tm * 16 * 68 + ks], 68, lane);
    const v16bf b0 = load_frag(&ldsVt[tn0 * 16 * 68 + ks], 68, lane);
    const v16bf b1 = load_frag(&ldsVt[(tn0 + 1) * 16 * 68 + ks], 68, lane);
    o0 = wmma_bf16(ap, b0, o0);
    o1 = wmma_bf16(ap, b1, o1);
  }
#pragma unroll
  for (int r = 0; r < 8; ++r) {
    const int l  = tm * 16 + r + 8 * half;
    const int hd = tn0 * 16 + n0;
    __bf16* dst = AO + ((size_t)(bb * Ln + l)) * (Hn * HDn) + h * HDn;
    dst[hd]      = (__bf16)o0[r];
    dst[hd + 16] = (__bf16)o1[r];
  }
}

// ---------------------------------------------------------------------------
// Kernel 3: output projection. attn[16384 x 1024] bf16 @ Wo[1024x1024] fp32
// -> fp32 out. grid = (8, 128).
// ---------------------------------------------------------------------------
__global__ __launch_bounds__(256) void outproj_kernel(
    const __bf16* __restrict__ A, const float* __restrict__ Wo,
    float* __restrict__ Out) {
  __shared__ __bf16 lds_a[128 * 34];
  __shared__ __bf16 lds_b[128 * 34];

  const int tid    = threadIdx.x;
  const int lane   = tid & 31;
  const int wv     = tid >> 5;
  const int wave_m = wv & 3;
  const int wave_n = wv >> 2;
  const int tileN  = blockIdx.x * 128;
  const int tileM  = blockIdx.y * 128;

  v8f acc[2][4];
#pragma unroll
  for (int tm = 0; tm < 2; ++tm)
#pragma unroll
    for (int tn = 0; tn < 4; ++tn) acc[tm][tn] = zero8();

  for (int kb = 0; kb < 32; ++kb) {
    if (kb + 1 < 32)
      __builtin_prefetch(A + (size_t)(tileM + (tid >> 3)) * Dn + (kb + 1) * 32, 0, 0);

#pragma unroll
    for (int i = 0; i < 4; ++i) {
      const int s   = tid + 256 * i;
      const int row = s >> 3;
      const int c4  = (s & 7) << 2;
      const bf16x4 a4 =
          *(const bf16x4*)(A + (size_t)(tileM + row) * Dn + kb * 32 + c4);
      __bf16* d = &lds_a[row * 34 + c4];
      d[0] = a4[0]; d[1] = a4[1]; d[2] = a4[2]; d[3] = a4[3];
    }
#pragma unroll
    for (int i = 0; i < 4; ++i) {
      const int s  = tid + 256 * i;
      const int k  = s >> 5;
      const int n4 = (s & 31) << 2;
      const float4 f =
          *(const float4*)(Wo + (size_t)(kb * 32 + k) * Dn + tileN + n4);
      lds_b[(n4 + 0) * 34 + k] = (__bf16)f.x;
      lds_b[(n4 + 1) * 34 + k] = (__bf16)f.y;
      lds_b[(n4 + 2) * 34 + k] = (__bf16)f.z;
      lds_b[(n4 + 3) * 34 + k] = (__bf16)f.w;
    }
    __syncthreads();

    const v16bf a0 = load_frag(&lds_a[(wave_m * 32 + 0)  * 34], 34, lane);
    const v16bf a1 = load_frag(&lds_a[(wave_m * 32 + 16) * 34], 34, lane);
#pragma unroll
    for (int tn = 0; tn < 4; ++tn) {
      const v16bf b = load_frag(&lds_b[(wave_n * 64 + tn * 16) * 34], 34, lane);
      acc[0][tn] = wmma_bf16(a0, b, acc[0][tn]);
      acc[1][tn] = wmma_bf16(a1, b, acc[1][tn]);
    }
    __syncthreads();
  }

  const int half = (lane >> 4) & 1;
  const int n0   = lane & 15;
#pragma unroll
  for (int tm = 0; tm < 2; ++tm)
#pragma unroll
    for (int tn = 0; tn < 4; ++tn)
#pragma unroll
      for (int r = 0; r < 8; ++r) {
        const int m = tileM + wave_m * 32 + tm * 16 + r + 8 * half;
        const int n = tileN + wave_n * 64 + tn * 16 + n0;
        Out[(size_t)m * Dn + n] = acc[tm][tn][r];
      }
}

// ---------------------------------------------------------------------------
// Host launcher
// ---------------------------------------------------------------------------
extern "C" void kernel_launch(void* const* d_in, const int* in_sizes, int n_in,
                              void* d_out, int out_size, void* d_ws,
                              size_t ws_size, hipStream_t stream) {
  const float* x    = (const float*)d_in[0];
  const float* bias = (const float*)d_in[1];
  const float* Wq   = (const float*)d_in[2];
  const float* Wk   = (const float*)d_in[3];
  const float* Wv   = (const float*)d_in[4];
  const float* Wo   = (const float*)d_in[5];
  float* out = (float*)d_out;

  // Workspace layout (bf16): q | k | v | attn_out, each 16384*1024 elements
  const size_t mat = (size_t)Mtot * Dn;   // 16,777,216 elements
  __bf16* ws = (__bf16*)d_ws;
  __bf16* qw = ws;
  __bf16* kw = ws + mat;
  __bf16* vw = ws + 2 * mat;
  __bf16* aw = ws + 3 * mat;

  qkv_kernel<<<dim3(24, 128), 256, 0, stream>>>(x, Wq, Wk, Wv, qw, kw, vw);
  attn_kernel<<<dim3(Bn * Hn), 256, 0, stream>>>(qw, kw, vw, bias, aw);
  outproj_kernel<<<dim3(8, 128), 256, 0, stream>>>(aw, Wo, out);
}
